// Cluster_14078902796308
// MI455X (gfx1250) — matrix operand order repaired and checked by reading, not compiled, not run on hardware
//
#include <hip/hip_runtime.h>

#define NS 32768
#define CC 1000
#define CPAD 1024
#define DD 1024
#define TOPK 16

typedef __attribute__((ext_vector_type(16))) __bf16 v16bf;
typedef __attribute__((ext_vector_type(8)))  __bf16 v8bf;
typedef __attribute__((ext_vector_type(4)))  __bf16 v4bf;
typedef __attribute__((ext_vector_type(8)))  float  v8f;
typedef __attribute__((ext_vector_type(4)))  float  v4f;

// ---------------------------------------------------------------------------
// Per-row 1/max(||f||,1e-8) for feature rows. One wave per row.
// ---------------------------------------------------------------------------
__global__ __launch_bounds__(256) void fnorm_kernel(const float* __restrict__ feature,
                                                    float* __restrict__ inv_fnorm) {
  const int wave = threadIdx.x >> 5, lane = threadIdx.x & 31;
  const int row = blockIdx.x * 8 + wave;
  const float* f = feature + (size_t)row * DD;
  float ss = 0.f;
  for (int j = lane * 4; j < DD; j += 128) {
    v4f x = *(const v4f*)(f + j);
    ss += x.x * x.x + x.y * x.y + x.z * x.z + x.w * x.w;
  }
  for (int off = 16; off >= 1; off >>= 1) ss += __shfl_xor(ss, off);
  if (lane == 0) inv_fnorm[row] = 1.0f / fmaxf(sqrtf(ss), 1e-8f);
}

// ---------------------------------------------------------------------------
// Centroid stats + bf16 pre-conversion (padded to 1024 rows, pad rows = 0 so
// the GEMM B-stage needs no bounds check and async-DMA can copy raw bytes).
//   yy[j]=||c_j||^2 (+inf pad -> never wins euclid argmin)
//   inv_cn[j]=1/max(||c_j||,1e-8) (NaN pad -> cos key NaN -> never wins argmax)
// ---------------------------------------------------------------------------
__global__ __launch_bounds__(256) void cstats_kernel(const float* __restrict__ centroids,
                                                     float* __restrict__ yy,
                                                     float* __restrict__ inv_cn,
                                                     __bf16* __restrict__ cb16) {
  const int wave = threadIdx.x >> 5, lane = threadIdx.x & 31;
  const int row = blockIdx.x * 8 + wave;   // grid = 128 -> 1024 waves
  if (row < CC) {
    const float* c = centroids + (size_t)row * DD;
    float ss = 0.f;
    for (int j = lane * 4; j < DD; j += 128) {
      v4f x = *(const v4f*)(c + j);
      ss += x.x * x.x + x.y * x.y + x.z * x.z + x.w * x.w;
      v4bf h = {(__bf16)x.x, (__bf16)x.y, (__bf16)x.z, (__bf16)x.w};
      *(v4bf*)(cb16 + (size_t)row * DD + j) = h;
    }
    for (int off = 16; off >= 1; off >>= 1) ss += __shfl_xor(ss, off);
    if (lane == 0) {
      yy[row] = ss;
      inv_cn[row] = 1.0f / fmaxf(sqrtf(ss), 1e-8f);
    }
  } else {
    v4bf z = {(__bf16)0.f, (__bf16)0.f, (__bf16)0.f, (__bf16)0.f};
    for (int j = lane * 4; j < DD; j += 128)
      *(v4bf*)(cb16 + (size_t)row * DD + j) = z;
    if (lane == 0) {
      yy[row] = __builtin_inff();
      inv_cn[row] = __builtin_nanf("");
    }
  }
}

// ---------------------------------------------------------------------------
// Max-softmax-prob per pred row = 1 / sum_j exp(p_j - max). One wave per row,
// single-pass online max/sum + butterfly combine.
// ---------------------------------------------------------------------------
__global__ __launch_bounds__(256) void scores_kernel(const float* __restrict__ pred,
                                                     float* __restrict__ scores) {
  const int wave = threadIdx.x >> 5, lane = threadIdx.x & 31;
  const int row = blockIdx.x * 8 + wave;
  const float* p = pred + (size_t)row * CC;
  float m = -__builtin_inff(), s = 0.f;
  for (int j = lane; j < CC; j += 32) {
    float v = p[j];
    if (v > m) { s = s * __expf(m - v) + 1.0f; m = v; }
    else        s += __expf(v - m);
  }
  for (int off = 16; off >= 1; off >>= 1) {
    float om = __shfl_xor(m, off), os = __shfl_xor(s, off);
    float M = fmaxf(m, om);
    s = s * __expf(m - M) + os * __expf(om - M);
    m = M;
  }
  if (lane == 0) scores[row] = 1.0f / s;
}

// ---------------------------------------------------------------------------
// bf16 WMMA GEMM (feature @ centroids^T) fused with per-row argmin(euclid key)
// and argmax(cos key), plus label computation.
//   Block: 256 threads = 8 waves (4 row-groups x 2 col-groups), tile 128x64.
//   Entire 128x1024 A slice staged in LDS as bf16 (uses CDNA5 320KB LDS).
//   B tiles DMA'd memory->LDS via global_load_async_to_lds_b128 (ASYNCcnt).
//   Wave tile: 32x32 = 2x2 v_wmma_f32_16x16x32_bf16 accumulators.
// ---------------------------------------------------------------------------
#define ASTRIDE 1032   // halfs: 1024 + 8 pad
#define BSTRIDE 136    // halfs: 128 + 8 pad

__global__ __launch_bounds__(256) void gemm_argminmax_kernel(
    const float* __restrict__ feature, const __bf16* __restrict__ cb16,
    const float* __restrict__ yy, const float* __restrict__ inv_cn,
    const float* __restrict__ inv_fnorm,
    int* __restrict__ label_idx, float* __restrict__ label_out) {
  extern __shared__ char smem[];
  __bf16* Alds = (__bf16*)smem;                                 // [128][ASTRIDE]
  __bf16* Blds = (__bf16*)(smem + 128 * ASTRIDE * 2);           // [64][BSTRIDE]

  const int t = threadIdx.x;
  const int lane = t & 31;
  const int wave = t >> 5;
  const int waveM = wave & 3;        // 0..3 -> 32-row group
  const int waveN = wave >> 2;       // 0..1 -> 32-col group
  const int half16 = lane >> 4;      // 0 or 1
  const int l16 = lane & 15;
  const int rowBlock = blockIdx.x * 128;

  // Stage full A slice (128 x 1024 f32 -> bf16), once per block.
  for (int i = t; i < 128 * DD / 4; i += 256) {
    const int idx = i * 4;
    const int r = idx >> 10;
    const int cL = idx & (DD - 1);
    v4f f = *(const v4f*)(feature + (size_t)(rowBlock + r) * DD + cL);
    v4bf h = {(__bf16)f.x, (__bf16)f.y, (__bf16)f.z, (__bf16)f.w};
    *(v4bf*)(Alds + r * ASTRIDE + cL) = h;
  }
  __syncthreads();

  float bestE[2][8], bestC[2][8];
  int   idxE[2][8],  idxC[2][8];
#pragma unroll
  for (int rt = 0; rt < 2; ++rt)
#pragma unroll
    for (int v = 0; v < 8; ++v) {
      bestE[rt][v] = __builtin_inff(); idxE[rt][v] = 0;
      bestC[rt][v] = -__builtin_inff(); idxC[rt][v] = 0;
    }

  for (int jb = 0; jb < CPAD / 64; ++jb) {
    const int jbase = jb * 64;
    v8f acc[2][2];
#pragma unroll
    for (int rt = 0; rt < 2; ++rt)
#pragma unroll
      for (int ct = 0; ct < 2; ++ct) acc[rt][ct] = (v8f)0.0f;

    for (int kc = 0; kc < DD; kc += 128) {
      // Async-DMA B tile (64 rows x 128 halfs, pre-padded bf16) into LDS:
      // 1024 16-byte chunks; each thread issues 4 global_load_async_to_lds.
#pragma unroll
      for (int q = 0; q < 4; ++q) {
        const int ci = q * 256 + t;
        const int r = ci >> 4;         // 0..63 centroid row in panel
        const int ch = ci & 15;        // 16B chunk within 256B row slice
        const unsigned lds_off =
            (unsigned)(size_t)(Blds + r * BSTRIDE + ch * 8);
        const __bf16* gp = cb16 + ((size_t)(jbase + r) * DD + kc + ch * 8);
        asm volatile("global_load_async_to_lds_b128 %0, %1, off"
                     :: "v"(lds_off), "v"((unsigned long long)(size_t)gp)
                     : "memory");
      }
      asm volatile("s_wait_asynccnt 0x0" ::: "memory");
      __syncthreads();

#pragma unroll
      for (int kk = 0; kk < 128; kk += 32) {
        // A fragments (16-bit A 16x32 layout: lanes<16 hold K 0..7 & 16..23,
        // lanes>=16 hold K 8..15 & 24..31; row = lane%16)
        v16bf afrag[2];
#pragma unroll
        for (int rt = 0; rt < 2; ++rt) {
          const int m = waveM * 32 + rt * 16 + l16;
          const __bf16* ap = Alds + m * ASTRIDE + kc + kk + half16 * 8;
          v8bf lo = *(const v8bf*)ap;
          v8bf hi = *(const v8bf*)(ap + 16);
          afrag[rt] = __builtin_shufflevector(lo, hi, 0, 1, 2, 3, 4, 5, 6, 7,
                                              8, 9, 10, 11, 12, 13, 14, 15);
        }
        // B fragments (16-bit B 32x16: col = lane%16, lanes<16 K 0..7 & 16..23,
        // lanes>=16 K 8..15 & 24..31)
        v16bf bfrag[2];
#pragma unroll
        for (int ct = 0; ct < 2; ++ct) {
          const int col = waveN * 32 + ct * 16 + l16;
          const __bf16* bp = Blds + col * BSTRIDE + kk + half16 * 8;
          v8bf lo = *(const v8bf*)bp;
          v8bf hi = *(const v8bf*)(bp + 16);
          bfrag[ct] = __builtin_shufflevector(lo, hi, 0, 1, 2, 3, 4, 5, 6, 7,
                                              8, 9, 10, 11, 12, 13, 14, 15);
        }
#pragma unroll
        for (int rt = 0; rt < 2; ++rt)
#pragma unroll
          for (int ct = 0; ct < 2; ++ct)
            acc[rt][ct] = __builtin_amdgcn_wmma_f32_16x16x32_bf16(
                false, afrag[rt], false, bfrag[ct], (short)0, acc[rt][ct],
                false, false);
      }
      __syncthreads();
    }

    // Fold this 64-column panel into per-lane running bests.
    // f32 C/D layout: VGPR v -> row (v + 8*half16), lane%16 -> column.
#pragma unroll
    for (int ct = 0; ct < 2; ++ct) {
      const int j = jbase + waveN * 32 + ct * 16 + l16;
      const float yv = yy[j];
      const float ic = inv_cn[j];
#pragma unroll
      for (int rt = 0; rt < 2; ++rt)
#pragma unroll
        for (int v = 0; v < 8; ++v) {
          const float d = acc[rt][ct][v];
          const float e = yv - 2.0f * d;   // euclid argmin key (xx const/row)
          if (e < bestE[rt][v]) { bestE[rt][v] = e; idxE[rt][v] = j; }
          const float ck = d * ic;         // cos argmax key (1/||f|| const/row)
          if (ck > bestC[rt][v]) { bestC[rt][v] = ck; idxC[rt][v] = j; }
        }
    }
  }

  // Cross-lane reduce within each 16-lane half (first-occurrence tie rules),
  // then lanes 0/16 emit label_idx + label.
#pragma unroll
  for (int rt = 0; rt < 2; ++rt)
#pragma unroll
    for (int v = 0; v < 8; ++v) {
      float e = bestE[rt][v], c = bestC[rt][v];
      int ie = idxE[rt][v], ic = idxC[rt][v];
      for (int off = 8; off >= 1; off >>= 1) {
        float oe = __shfl_xor(e, off);  int oie = __shfl_xor(ie, off);
        float oc = __shfl_xor(c, off);  int oic = __shfl_xor(ic, off);
        if (oe < e || (oe == e && oie < ie)) { e = oe; ie = oie; }
        if (oc > c || (oc == c && oic < ic)) { c = oc; ic = oic; }
      }
      if (l16 == 0) {
        const int row = rowBlock + waveM * 32 + rt * 16 + v + 8 * half16;
        label_idx[row] = ic;
        const float cosmax = c * inv_fnorm[row];
        label_out[row] = (ic == ie && cosmax > 0.85f) ? (float)ic : -1.0f;
      }
    }
}

// ---------------------------------------------------------------------------
// Sequential bounded top-K scatter, parallelized over classes (updates to
// different classes are independent; per-class item order preserved).
// One wave per class; lanes 0..15 hold the 16-slot row in registers.
// ---------------------------------------------------------------------------
__global__ __launch_bounds__(32) void topk_scan_kernel(
    const int* __restrict__ label_idx, const float* __restrict__ scores,
    const int* __restrict__ uindex, const float* __restrict__ ptn_in,
    const int* __restrict__ fidx_in, float* __restrict__ ptn_out,
    float* __restrict__ fidx_out) {
  const int c = blockIdx.x;
  const int lane = threadIdx.x;
  float rowv = (lane < TOPK) ? ptn_in[c * TOPK + lane] : 0.f;
  int fi = (lane < TOPK) ? fidx_in[c * TOPK + lane] : 0;

  for (int base = 0; base < NS; base += 32) {
    const int li = label_idx[base + lane];
    const bool match = (li == c);
    float sc = 0.f;
    int ui = 0;
    if (match) { sc = scores[base + lane]; ui = uindex[base + lane]; }
    unsigned mask = (unsigned)__ballot(match);
    while (mask) {
      const int l = __ffs(mask) - 1;
      mask &= mask - 1;
      const float s = __shfl(sc, l);
      const int u = __shfl(ui, l);
      float mn = (lane < TOPK) ? rowv : __builtin_inff();
      for (int off = 16; off >= 1; off >>= 1) mn = fminf(mn, __shfl_xor(mn, off));
      const unsigned eq = (unsigned)__ballot(lane < TOPK && rowv == mn);
      const int mi = __ffs(eq) - 1;                 // first min (jnp.argmin)
      if (lane == mi && s > mn) { rowv = s; fi = u; }
    }
  }
  if (lane < TOPK) {
    ptn_out[c * TOPK + lane] = rowv;
    fidx_out[c * TOPK + lane] = (float)fi;
  }
}

// ---------------------------------------------------------------------------
extern "C" void kernel_launch(void* const* d_in, const int* in_sizes, int n_in,
                              void* d_out, int out_size, void* d_ws, size_t ws_size,
                              hipStream_t stream) {
  const float* feature   = (const float*)d_in[0];
  const float* pred      = (const float*)d_in[1];
  const int*   uindex    = (const int*)d_in[2];
  const float* centroids = (const float*)d_in[3];
  const float* ptn_in    = (const float*)d_in[4];
  const int*   fidx_in   = (const int*)d_in[5];

  char* ws = (char*)d_ws;
  float* inv_fnorm = (float*)ws;              ws += (size_t)NS * 4;
  float* scores    = (float*)ws;              ws += (size_t)NS * 4;
  int*   label_idx = (int*)ws;                ws += (size_t)NS * 4;
  float* yy        = (float*)ws;              ws += (size_t)CPAD * 4;
  float* inv_cn    = (float*)ws;              ws += (size_t)CPAD * 4;
  __bf16* cb16     = (__bf16*)ws;             ws += (size_t)CPAD * DD * 2;

  float* label_out = (float*)d_out;                   // [NS]
  float* ptn_out   = label_out + NS;                  // [CC*TOPK]
  float* fidx_out  = ptn_out + (size_t)CC * TOPK;     // [CC*TOPK]

  const int smem_bytes = 128 * ASTRIDE * 2 + 64 * BSTRIDE * 2;  // 281600 B
  (void)hipFuncSetAttribute((const void*)gemm_argminmax_kernel,
                            hipFuncAttributeMaxDynamicSharedMemorySize,
                            smem_bytes);

  fnorm_kernel<<<NS / 8, 256, 0, stream>>>(feature, inv_fnorm);
  cstats_kernel<<<CPAD / 8, 256, 0, stream>>>(centroids, yy, inv_cn, cb16);
  scores_kernel<<<NS / 8, 256, 0, stream>>>(pred, scores);
  gemm_argminmax_kernel<<<NS / 128, 256, smem_bytes, stream>>>(
      feature, cb16, yy, inv_cn, inv_fnorm, label_idx, label_out);
  topk_scan_kernel<<<CC, 32, 0, stream>>>(label_idx, scores, uindex, ptn_in,
                                          fidx_in, ptn_out, fidx_out);
}
